// GATNet_1_1090921693355
// MI455X (gfx1250) — matrix operand order repaired and checked by reading, not compiled, not run on hardware
//
#include <hip/hip_runtime.h>
#include <hip/hip_bf16.h>

typedef __attribute__((ext_vector_type(16))) _Float16 v16h;
typedef __attribute__((ext_vector_type(8)))  float    v8f;

#define NN 50000
#define EE 800000
#define HH 3
#define DD 64
#define GG 1024

// ---------------- pack f32 A [M,K] -> f16 WMMA A-fragment layout ----------------
// Per (16x32) tile: 512 f16, lane-major (32 lanes x 16 elems), elems in VGPR order.
// A fragment (ISA 7.12.2, 16-bit A 16x32): lanes<16: K=0..7 (v0..3), 16..23 (v4..7);
// lanes>=16: K=8..15, 24..31.  kr = ((v&4)<<2) + (half<<3) + ((v&3)<<1) + low
__global__ void k_pack_a(const float* __restrict__ in, _Float16* __restrict__ out,
                         int K, long ntot) {
  long o = (long)blockIdx.x * blockDim.x + threadIdx.x;
  if (o >= ntot) return;
  int tileIdx = (int)(o >> 9);
  int r       = (int)(o & 511);
  int lane = r >> 4, j = r & 15;
  int kblocks = K >> 5;
  int tm = tileIdx / kblocks, kb = tileIdx - tm * kblocks;
  int half = lane >> 4, lm = lane & 15;
  int v = j >> 1, low = j & 1;
  int kr = ((v & 4) << 2) + (half << 3) + ((v & 3) << 1) + low;
  out[o] = (_Float16)in[(size_t)(tm * 16 + lm) * K + kb * 32 + kr];
}

// ---------------- pack f32 B [K,Ncol] -> f16 WMMA B-fragment layout ----------------
// B fragment (32x16 f16): lanes<16 hold K=0..7,16..23, lanes>=16 hold K=8..15,24..31; N=lane%16.
// kr = ((j&8)<<1) + (half<<3) + (j&7).  Tile order: (kb, tn) -> (kb*tilesN + tn)*512.
__global__ void k_pack_b(const float* __restrict__ in, _Float16* __restrict__ out,
                         int Ncol, long ntot) {
  long o = (long)blockIdx.x * blockDim.x + threadIdx.x;
  if (o >= ntot) return;
  int tileIdx = (int)(o >> 9);
  int r       = (int)(o & 511);
  int lane = r >> 4, j = r & 15;
  int tilesN = Ncol >> 4;
  int kb = tileIdx / tilesN, tn = tileIdx - kb * tilesN;
  int half = lane >> 4, lm = lane & 15;
  int kr = ((j & 8) << 1) + (half << 3) + (j & 7);
  out[o] = (_Float16)in[(size_t)(kb * 32 + kr) * Ncol + tn * 16 + lm];
}

// ---------------- WMMA GEMM on pre-packed fragments ----------------
// C[M,Ncol] = A[M,K] @ B[K,Ncol] + bias ; one wave per 16x16 tile.
// Inner loop: 2x b128 loads (A) + 2x b128 loads (B) + 1x v_wmma_f32_16x16x32_f16.
__global__ void k_wmma_gemm_f16(const _Float16* __restrict__ Apack,
                                const _Float16* __restrict__ Bpack,
                                const float* __restrict__ bias,
                                float* __restrict__ C,
                                int K, int Ncol, int totalTiles) {
  int wave = blockIdx.x * (blockDim.x >> 5) + (threadIdx.x >> 5);
  if (wave >= totalTiles) return;              // wave-uniform: EXEC stays all-ones
  int tilesPerRow = Ncol >> 4;
  int tm = wave / tilesPerRow;
  int tn = wave - tm * tilesPerRow;
  int lane = threadIdx.x & 31;
  int hf   = lane >> 4;
  int lm   = lane & 15;
  int kblocks = K >> 5;

  const _Float16* ap = Apack + ((size_t)tm * kblocks) * 512 + lane * 16;
  const _Float16* bp = Bpack + (size_t)tn * 512 + lane * 16;
  size_t bstride = (size_t)tilesPerRow * 512;

  v16h a, b;
  v8f c = {};
  for (int kb = 0; kb < kblocks; ++kb) {
    a = *(const v16h*)ap;                      // 32 contiguous bytes/lane
    b = *(const v16h*)bp;
    ap += 512;
    bp += bstride;
    c = __builtin_amdgcn_wmma_f32_16x16x32_f16(false, a, false, b,
                                               (short)0, c, false, false);
  }

  float bv = bias ? bias[tn * 16 + lm] : 0.0f;
  int mb = tm * 16 + (hf << 3);
#pragma unroll
  for (int r = 0; r < 8; ++r)
    C[(size_t)(mb + r) * Ncol + tn * 16 + lm] = c[r] + bv;
}

// ---------------- per-(node,head) attention logits + init of max/denom ----------------
__global__ void k_el_er(const float* __restrict__ z, const float* __restrict__ al,
                        const float* __restrict__ ar, float* __restrict__ el,
                        float* __restrict__ er, float* __restrict__ mx,
                        float* __restrict__ dn, int nh) {
  int i = blockIdx.x * blockDim.x + threadIdx.x;
  if (i >= nh) return;
  int n = i / HH, h = i - n * HH;
  const float* zp  = z  + (size_t)n * (HH * DD) + h * DD;
  const float* alp = al + h * DD;
  const float* arp = ar + h * DD;
  float sl = 0.f, sr = 0.f;
#pragma unroll 4
  for (int d = 0; d < DD; ++d) { float zv = zp[d]; sl += zv * alp[d]; sr += zv * arp[d]; }
  el[i] = sl; er[i] = sr; mx[i] = -3.0e38f; dn[i] = 0.f;
}

// ---------------- out[n, c] = bias[c] (accumulation base) ----------------
__global__ void k_init_out(float* __restrict__ out, const float* __restrict__ b, long n) {
  long i = (long)blockIdx.x * blockDim.x + threadIdx.x;
  if (i < n) out[i] = b[i % (HH * DD)];
}

__device__ inline void atomicMaxFloat(float* addr, float val) {
  unsigned int* ua = (unsigned int*)addr;
  unsigned int old = *ua;
  while (__uint_as_float(old) < val) {
    unsigned int assumed = old;
    old = atomicCAS(ua, assumed, __float_as_uint(val));
    if (old == assumed) break;
  }
}

// ---------------- edge pass 1: e = leaky_relu(el[src]+er[dst]); segment max ----------------
__global__ void k_edge1(const int* __restrict__ src, const int* __restrict__ dst,
                        const float* __restrict__ el, const float* __restrict__ er,
                        float* __restrict__ ebuf, float* __restrict__ mx, long n) {
  long i = (long)blockIdx.x * blockDim.x + threadIdx.x;
  if (i >= n) return;
  long e = i / HH; int h = (int)(i - e * HH);
  int s = src[e], d = dst[e];
  float v = el[s * HH + h] + er[d * HH + h];
  v = (v > 0.f) ? v : 0.2f * v;
  ebuf[i] = v;
  atomicMaxFloat(&mx[d * HH + h], v);
}

// ---------------- edge pass 2: ex = exp(e - m[dst]); segment sum ----------------
__global__ void k_edge2(const int* __restrict__ dst, float* __restrict__ ebuf,
                        const float* __restrict__ mx, float* __restrict__ dn, long n) {
  long i = (long)blockIdx.x * blockDim.x + threadIdx.x;
  if (i >= n) return;
  long e = i / HH; int h = (int)(i - e * HH);
  int d = dst[e];
  float ex = __expf(ebuf[i] - mx[d * HH + h]);
  ebuf[i] = ex;
  atomicAdd(&dn[d * HH + h], ex);
}

// ---------------- edge pass 3: alpha = ex / denom[dst] ----------------
__global__ void k_alpha(const int* __restrict__ dst, float* __restrict__ ebuf,
                        const float* __restrict__ dn, long n) {
  long i = (long)blockIdx.x * blockDim.x + threadIdx.x;
  if (i >= n) return;
  long e = i / HH; int h = (int)(i - e * HH);
  ebuf[i] = ebuf[i] / dn[dst[e] * HH + h];
}

// ---------------- edge pass 4: one block (192 threads) per edge, coalesced ----------------
__global__ void k_edge_agg(const int* __restrict__ src, const int* __restrict__ dst,
                           const float* __restrict__ ebuf, const float* __restrict__ z,
                           float* __restrict__ out) {
  int e = blockIdx.x;
  int c = threadIdx.x;                 // 0..191
  int h = c >> 6;
  int s = src[e], d = dst[e];
  float w = ebuf[(size_t)e * HH + h];
  atomicAdd(&out[(size_t)d * (HH * DD) + c], z[(size_t)s * (HH * DD) + c] * w);
}

// ---------------- pooling ----------------
__global__ void k_zero(float* __restrict__ p, long n) {
  long i = (long)blockIdx.x * blockDim.x + threadIdx.x;
  if (i < n) p[i] = 0.f;
}
__global__ void k_pool(const float* __restrict__ x, const int* __restrict__ gid,
                       float* __restrict__ sums, float* __restrict__ cnts, long n) {
  long i = (long)blockIdx.x * blockDim.x + threadIdx.x;
  if (i >= n) return;
  long nd = i / DD; int d = (int)(i - nd * DD);
  int g = gid[nd];
  atomicAdd(&sums[(size_t)g * DD + d], x[i]);
  if (d == 0) atomicAdd(&cnts[g], 1.0f);
}
__global__ void k_final(const float* __restrict__ sums, const float* __restrict__ cnts,
                        float* __restrict__ out, long n) {
  long i = (long)blockIdx.x * blockDim.x + threadIdx.x;
  if (i >= n) return;
  out[i] = sums[i] / fmaxf(cnts[i / DD], 1.0f);
}

static inline unsigned cdivu(long a, long b) { return (unsigned)((a + b - 1) / b); }

extern "C" void kernel_launch(void* const* d_in, const int* in_sizes, int n_in,
                              void* d_out, int out_size, void* d_ws, size_t ws_size,
                              hipStream_t stream) {
  (void)in_sizes; (void)n_in; (void)out_size; (void)ws_size;
  const float* feats = (const float*)d_in[0];
  const int*   src   = (const int*)d_in[1];
  const int*   dst   = (const int*)d_in[2];
  const int*   gid   = (const int*)d_in[3];
  const float* W1  = (const float*)d_in[4];
  const float* al1 = (const float*)d_in[5];
  const float* ar1 = (const float*)d_in[6];
  const float* b1  = (const float*)d_in[7];
  const float* W2  = (const float*)d_in[8];
  const float* al2 = (const float*)d_in[9];
  const float* ar2 = (const float*)d_in[10];
  const float* b2  = (const float*)d_in[11];
  const float* Wl  = (const float*)d_in[12];
  const float* bl  = (const float*)d_in[13];

  // ---- workspace bump allocator (256B aligned) ----
  char* p = (char*)d_ws;
  auto alloc = [&](size_t bytes) -> char* {
    char* r = p; p += (bytes + 255) & ~(size_t)255; return r;
  };
  float*     z    = (float*)    alloc((size_t)NN * HH * DD * 4);
  float*     outb = (float*)    alloc((size_t)NN * HH * DD * 4);
  float*     x    = (float*)    alloc((size_t)NN * DD * 4);
  _Float16*  Ap   = (_Float16*) alloc((size_t)NN * HH * DD * 2);   // packed A (max N*192)
  float*     ebuf = (float*)    alloc((size_t)EE * HH * 4);
  float*     el   = (float*)    alloc((size_t)NN * HH * 4);
  float*     er   = (float*)    alloc((size_t)NN * HH * 4);
  float*     mx   = (float*)    alloc((size_t)NN * HH * 4);
  float*     dn   = (float*)    alloc((size_t)NN * HH * 4);
  _Float16*  W1p  = (_Float16*) alloc((size_t)128 * HH * DD * 2);  // packed B
  _Float16*  W2p  = (_Float16*) alloc((size_t)DD * HH * DD * 2);
  _Float16*  Wlp  = (_Float16*) alloc((size_t)HH * DD * DD * 2);
  float*     sums = (float*)    alloc((size_t)GG * DD * 4);
  float*     cnts = (float*)    alloc((size_t)GG * 4);

  const unsigned B = 256;
  // weight fragment packing (tiny, once per call)
  k_pack_b<<<cdivu(128L * HH * DD, B), B, 0, stream>>>(W1, W1p, HH * DD, 128L * HH * DD);
  k_pack_b<<<cdivu((long)DD * HH * DD, B), B, 0, stream>>>(W2, W2p, HH * DD, (long)DD * HH * DD);
  k_pack_b<<<cdivu((long)HH * DD * DD, B), B, 0, stream>>>(Wl, Wlp, DD, (long)HH * DD * DD);

  long NHD = (long)NN * HH * DD;
  long EH  = (long)EE * HH;

  auto conv = [&](const float* hin, int Fin, const _Float16* Wp,
                  const float* al, const float* ar, const float* b) {
    k_pack_a<<<cdivu((long)NN * Fin, B), B, 0, stream>>>(hin, Ap, Fin, (long)NN * Fin);
    int tiles = (NN / 16) * ((HH * DD) / 16);               // 3125 * 12
    k_wmma_gemm_f16<<<cdivu(tiles, 4), 128, 0, stream>>>(Ap, Wp, nullptr, z,
                                                         Fin, HH * DD, tiles);
    k_el_er<<<cdivu((long)NN * HH, B), B, 0, stream>>>(z, al, ar, el, er, mx, dn, NN * HH);
    k_init_out<<<cdivu(NHD, B), B, 0, stream>>>(outb, b, NHD);
    k_edge1<<<cdivu(EH, B), B, 0, stream>>>(src, dst, el, er, ebuf, mx, EH);
    k_edge2<<<cdivu(EH, B), B, 0, stream>>>(dst, ebuf, mx, dn, EH);
    k_alpha<<<cdivu(EH, B), B, 0, stream>>>(dst, ebuf, dn, EH);
    k_edge_agg<<<EE, HH * DD, 0, stream>>>(src, dst, ebuf, z, outb);
  };
  auto linear = [&]() {   // x = outb @ Wl + bl
    k_pack_a<<<cdivu(NHD, B), B, 0, stream>>>(outb, Ap, HH * DD, NHD);
    int tiles = (NN / 16) * (DD / 16);                      // 3125 * 4
    k_wmma_gemm_f16<<<cdivu(tiles, 4), 128, 0, stream>>>(Ap, Wlp, bl, x,
                                                         HH * DD, DD, tiles);
  };

  conv(feats, 128, W1p, al1, ar1, b1); linear();
  conv(x, DD, W2p, al2, ar2, b2);      linear();   // LeakyReLU(slope=1) == identity
  conv(x, DD, W2p, al2, ar2, b2);      linear();

  // per-graph mean pooling
  k_zero<<<cdivu((long)GG * DD + GG, B), B, 0, stream>>>(sums, (long)GG * DD + GG); // sums+cnts contiguous
  k_pool<<<cdivu((long)NN * DD, B), B, 0, stream>>>(x, gid, sums, cnts, (long)NN * DD);
  k_final<<<cdivu((long)GG * DD, B), B, 0, stream>>>(sums, cnts, (float*)d_out, (long)GG * DD);
}
